// SimpleNeuroSAT_12618613916111
// MI455X (gfx1250) — matrix operand order repaired and checked by reading, not compiled, not run on hardware
//
#include <hip/hip_runtime.h>
#include <hip/hip_bf16.h>

// ---------------------------------------------------------------------------
// SimpleNeuroSAT forward on gfx1250: bf16 WMMA GEMMs + f32 edge atomics.
// All GEMM N dims padded to multiples of 64 -> branch-free 4-tile inner loop.
// ---------------------------------------------------------------------------

typedef __bf16 bf16_t;
typedef bf16_t v16bf __attribute__((ext_vector_type(16)));
typedef bf16_t v8bf  __attribute__((ext_vector_type(8)));
typedef float  v8f   __attribute__((ext_vector_type(8)));

#define NVARS    8000
#define NLITS    16000
#define NCLAUSES 34000
#define NEDGES   102000
#define NGRAPHS  16
#define FMDIM    128
#define NROUNDS  8

#define CDIV(a,b) (((a)+(b)-1)/(b))

// ------------------------------ small utils --------------------------------
__global__ void k_zero(float* __restrict__ p, int n) {
  int i = blockIdx.x * blockDim.x + threadIdx.x;
  if (i < n) p[i] = 0.0f;
}

__global__ void k_fill_scalar(float* __restrict__ p, int n, const float* __restrict__ s) {
  int i = blockIdx.x * blockDim.x + threadIdx.x;
  if (i < n) p[i] = *s;
}

__device__ __forceinline__ float softplusf(float x) {
  return (x > 20.0f) ? x : log1pf(expf(x));
}

// ------------------------- weight fragment packing -------------------------
// Pack f32 row-major W[Kreal,Nreal] into bf16 WMMA B-fragments, zero-padded
// to [Nt x Kt x (32 lanes x 16 elems)].
// B layout (16-bit, 32x16): lane<16: col=lane&15, K=kt*32 + {0..7, 16..23}
//                           lane>=16: col=lane&15, K=kt*32 + {8..15, 24..31}
__global__ void k_prep_wfrag(const float* __restrict__ W, bf16_t* __restrict__ dst,
                             int Kreal, int Nreal, int Kt, int Nt) {
  int idx = blockIdx.x * blockDim.x + threadIdx.x;
  int total = Nt * Kt * 512;
  if (idx >= total) return;
  int j    = idx & 15;
  int lane = (idx >> 4) & 31;
  int kt   = (idx >> 9) % Kt;
  int t    = (idx >> 9) / Kt;
  int col  = t * 16 + (lane & 15);
  int k    = kt * 32 + ((lane >> 4) * 8) + (j & 7) + ((j >> 3) * 16);
  float v = (k < Kreal && col < Nreal) ? W[(size_t)k * Nreal + col] : 0.0f;
  dst[idx] = (bf16_t)v;
}

__global__ void k_prep_bias(const float* __restrict__ b, float* __restrict__ dst,
                            int Nreal, int Np) {
  int i = blockIdx.x * blockDim.x + threadIdx.x;
  if (i < Np) dst[i] = (i < Nreal) ? b[i] : 0.0f;
}

// ------------------------------- WMMA GEMM ---------------------------------
// Y[M,Np] = act(X[M,Kp](bf16) @ Wfrag + bias). One wave per 16-row M-tile,
// 4 full N-tiles (64 cols) per group (Np % 64 == 0). act: 0=linear, 1=relu6.
__global__ void __launch_bounds__(128)
k_gemm_wmma(const bf16_t* __restrict__ X, const bf16_t* __restrict__ Wf,
            const float* __restrict__ bias,
            float* __restrict__ Yf32, bf16_t* __restrict__ Ybf16,
            int M, int Kp, int Np, int act) {
  const int wave = threadIdx.x >> 5;
  const int lane = threadIdx.x & 31;
  const int mtile = blockIdx.x * 4 + wave;
  if (mtile * 16 >= M) return;

  const int r16 = lane & 15;
  const int hi  = lane >> 4;        // 0 or 1
  const int kgrp = hi * 8;          // A-fragment K sub-group
  const int row  = mtile * 16 + r16;
  const int Kt = Kp >> 5;
  const int Nq = Np >> 6;           // groups of 4 N-tiles (always exact)

  const bf16_t* xrow = X + (size_t)row * Kp + kgrp;
  const size_t tstride = (size_t)Kt * 512;   // elements per N-tile of Wf

  const v8f vzero = v8f{0.f,0.f,0.f,0.f,0.f,0.f,0.f,0.f};

  for (int nq = 0; nq < Nq; ++nq) {
    v8f acc0 = vzero, acc1 = vzero, acc2 = vzero, acc3 = vzero;
    const bf16_t* wq = Wf + (size_t)(nq * 4) * tstride + (size_t)lane * 16;

    for (int kt = 0; kt < Kt; ++kt) {
      // A fragment: lane<16 holds K kgrp+0..7 and kgrp+16..23 of its row.
      const bf16_t* xp = xrow + kt * 32;
      v8bf alo = *(const v8bf*)(xp);
      v8bf ahi = *(const v8bf*)(xp + 16);
      v16bf a = __builtin_shufflevector(alo, ahi,
                  0,1,2,3,4,5,6,7, 8,9,10,11,12,13,14,15);
      // Four B fragments (contiguous 32B per lane, coalesced across wave).
      const bf16_t* wp = wq + (size_t)kt * 512;
      v16bf b0 = *(const v16bf*)(wp);
      v16bf b1 = *(const v16bf*)(wp + tstride);
      v16bf b2 = *(const v16bf*)(wp + 2 * tstride);
      v16bf b3 = *(const v16bf*)(wp + 3 * tstride);
      acc0 = __builtin_amdgcn_wmma_f32_16x16x32_bf16(false, a, false, b0, (short)0, acc0, false, false);
      acc1 = __builtin_amdgcn_wmma_f32_16x16x32_bf16(false, a, false, b1, (short)0, acc1, false, false);
      acc2 = __builtin_amdgcn_wmma_f32_16x16x32_bf16(false, a, false, b2, (short)0, acc2, false, false);
      acc3 = __builtin_amdgcn_wmma_f32_16x16x32_bf16(false, a, false, b3, (short)0, acc3, false, false);
    }

    // D layout: VGPR v, lane<16 -> row mtile*16+v, lane>=16 -> row mtile*16+8+v
    const int rbase = mtile * 16 + hi * 8;
    v8f accs[4] = {acc0, acc1, acc2, acc3};
    #pragma unroll
    for (int i = 0; i < 4; ++i) {
      const int col = (nq * 4 + i) * 16 + r16;
      const float bc = bias[col];
      #pragma unroll
      for (int v = 0; v < 8; ++v) {
        float val = accs[i][v] + bc;
        if (act) val = fminf(fmaxf(val, 0.0f), 6.0f);
        size_t offo = (size_t)(rbase + v) * Np + col;
        if (Yf32) Yf32[offo] = val;
        else      Ybf16[offo] = (bf16_t)val;
      }
    }
  }
}

// ------------------------------ edge kernels -------------------------------
__global__ void k_edge_cv(const int* __restrict__ rows, const int* __restrict__ cols,
                          const float* __restrict__ lgt, float* __restrict__ cv) {
  int e = blockIdx.x * blockDim.x + threadIdx.x;
  if (e >= NEDGES) return;
  int r = rows[e], c = cols[e];
  float lg = (r < NVARS) ? lgt[r] : -lgt[r - NVARS];
  atomicAdd(&cv[c], softplusf(lg));
}

__global__ void k_edge_lc(const int* __restrict__ rows, const int* __restrict__ cols,
                          const float* __restrict__ L, float* __restrict__ LC) {
  int idx = blockIdx.x * blockDim.x + threadIdx.x;
  if (idx >= NEDGES * 64) return;
  int e = idx >> 6, f = idx & 63;
  int r = rows[e], c = cols[e];
  float v = (r < NVARS) ? L[(size_t)r * FMDIM + f]
                        : L[(size_t)(r - NVARS) * FMDIM + 64 + f];
  atomicAdd(&LC[(size_t)c * 64 + f], v);
}

__global__ void k_edge_cl(const int* __restrict__ rows, const int* __restrict__ cols,
                          const float* __restrict__ C2, float* __restrict__ CL) {
  int idx = blockIdx.x * blockDim.x + threadIdx.x;
  if (idx >= NEDGES * 128) return;
  int e = idx >> 7, f = idx & 127;
  int r = rows[e], c = cols[e];
  atomicAdd(&CL[(size_t)r * FMDIM + f], C2[(size_t)c * FMDIM + f]);
}

// ----------------------------- input assembly ------------------------------
__global__ void k_assemble_xc(const float* __restrict__ C, const float* __restrict__ cv,
                              const float* __restrict__ LC, const float* __restrict__ lc_scale,
                              bf16_t* __restrict__ Xc) {
  int idx = blockIdx.x * blockDim.x + threadIdx.x;
  if (idx >= NCLAUSES * 224) return;
  int r = idx / 224, c = idx % 224;
  float v;
  if (c < 128)      v = C[(size_t)r * FMDIM + c];
  else if (c == 128) v = expf(-cv[r]);
  else if (c < 193)  v = LC[(size_t)r * 64 + (c - 129)] * (*lc_scale);
  else               v = 0.0f;
  Xc[idx] = (bf16_t)v;
}

__global__ void k_assemble_xl(const float* __restrict__ L, const float* __restrict__ CL,
                              const float* __restrict__ cl_scale, bf16_t* __restrict__ Xl) {
  int idx = blockIdx.x * blockDim.x + threadIdx.x;
  if (idx >= NVARS * 384) return;
  int r = idx / 384, c = idx % 384;
  float v;
  if (c < 128)      v = L[(size_t)r * FMDIM + c];
  else if (c < 256)  v = CL[(size_t)r * FMDIM + (c - 128)] * (*cl_scale);
  else               v = CL[(size_t)(r + NVARS) * FMDIM + (c - 256)] * (*cl_scale);
  Xl[idx] = (bf16_t)v;
}

// ------------------------------ normalization ------------------------------
__global__ void k_colsum(const float* __restrict__ X, float* __restrict__ s,
                         float* __restrict__ sq, int M, int N) {
  int c = threadIdx.x;                 // blockDim.x == N (128)
  int r0 = blockIdx.x * 512;
  int r1 = (r0 + 512 < M) ? (r0 + 512) : M;
  float a = 0.0f, b = 0.0f;
  for (int r = r0; r < r1; ++r) {
    float x = X[(size_t)r * N + c];
    a += x; b += x * x;
  }
  atomicAdd(&s[c], a);
  atomicAdd(&sq[c], b);
}

__global__ void k_norm_apply(float* __restrict__ X, const float* __restrict__ s,
                             const float* __restrict__ sq, int M, int N,
                             bf16_t* __restrict__ ob) {
  int i = blockIdx.x * blockDim.x + threadIdx.x;
  if (i >= M * N) return;
  int c = i % N;
  float mean = s[c] / (float)M;
  float var  = sq[c] / (float)M - mean * mean;
  float y = (X[i] - mean) * rsqrtf(var + 1e-6f);
  X[i] = y;
  if (ob) ob[i] = (bf16_t)y;
}

// ------------------------------ loss + output ------------------------------
__global__ void k_extract_logits(const float* __restrict__ lp, float* __restrict__ lg) {
  int i = blockIdx.x * blockDim.x + threadIdx.x;
  if (i < NVARS) lg[i] = lp[(size_t)i * 64];
}

__global__ void k_clause_loss(const float* __restrict__ cv, const int* __restrict__ gid,
                              float* __restrict__ pg) {
  int c = blockIdx.x * blockDim.x + threadIdx.x;
  if (c >= NCLAUSES) return;
  float ecv = expf(-cv[c]);
  float pc = ecv * (-logf(1.0f - ecv + 1e-8f));
  atomicAdd(&pg[gid[c]], pc);
}

__global__ void k_loss_accum(const float* __restrict__ pg, float* __restrict__ loss) {
  if (threadIdx.x == 0 && blockIdx.x == 0) {
    float t = 0.0f;
    for (int g = 0; g < NGRAPHS; ++g) t += sqrtf(pg[g] + 1e-6f);
    *loss += t;
  }
}

__global__ void k_finalize(const float* __restrict__ lg, const float* __restrict__ loss,
                           float* __restrict__ out) {
  int i = blockIdx.x * blockDim.x + threadIdx.x;
  if (i < NVARS) out[i] = lg[i];
  else if (i == NVARS) out[i] = (*loss) / (float)NROUNDS;
}

// ---------------------------------------------------------------------------
extern "C" void kernel_launch(void* const* d_in, const int* in_sizes, int n_in,
                              void* d_out, int out_size, void* d_ws, size_t ws_size,
                              hipStream_t stream) {
  // Input order: top-level dict insertion order; params flattened with jax
  // tree order (sorted keys): CL_scale, C_init_scale, C_updates(W,b)x3,
  // LC_scale, L_init_scale, L_updates(W,b)x3, V_score(W,b)x3.
  const int*   adj_rows = (const int*)d_in[0];
  const int*   adj_cols = (const int*)d_in[1];
  const int*   gids     = (const int*)d_in[2];
  const float* cl_scale = (const float*)d_in[3];
  const float* c_init   = (const float*)d_in[4];
  const float* lc_scale = (const float*)d_in[11];
  const float* l_init   = (const float*)d_in[12];

  struct Layer { const float* W; const float* B; int Kr, Nr, Kp, Np; };
  Layer Ls[9] = {
    {(const float*)d_in[5],  (const float*)d_in[6],  193, 256, 224, 256}, // C_up 0
    {(const float*)d_in[7],  (const float*)d_in[8],  256, 256, 256, 256}, // C_up 1
    {(const float*)d_in[9],  (const float*)d_in[10], 256, 128, 256, 128}, // C_up 2
    {(const float*)d_in[13], (const float*)d_in[14], 384, 384, 384, 384}, // L_up 0
    {(const float*)d_in[15], (const float*)d_in[16], 384, 384, 384, 384}, // L_up 1
    {(const float*)d_in[17], (const float*)d_in[18], 384, 128, 384, 128}, // L_up 2
    {(const float*)d_in[19], (const float*)d_in[20], 128, 256, 128, 256}, // V 0
    {(const float*)d_in[21], (const float*)d_in[22], 256, 256, 256, 256}, // V 1
    {(const float*)d_in[23], (const float*)d_in[24], 256,   1, 256,  64}, // V 2 (N pad 1->64)
  };

  // ---- workspace carving --------------------------------------------------
  size_t off = 0;
  auto carve = [&](size_t bytes) -> void* {
    void* p = (void*)((char*)d_ws + off);
    off += (bytes + 255) & ~(size_t)255;
    return p;
  };
  float*  L_a  = (float*)carve((size_t)NVARS * FMDIM * 4);
  float*  L_b  = (float*)carve((size_t)NVARS * FMDIM * 4);
  float*  C_a  = (float*)carve((size_t)NCLAUSES * FMDIM * 4);
  float*  C_b  = (float*)carve((size_t)NCLAUSES * FMDIM * 4);
  bf16_t* Xc   = (bf16_t*)carve((size_t)NCLAUSES * 224 * 2);
  bf16_t* SA   = (bf16_t*)carve((size_t)NCLAUSES * 256 * 2);
  bf16_t* SB   = (bf16_t*)carve((size_t)NCLAUSES * 256 * 2);
  float*  LC   = (float*)carve((size_t)NCLAUSES * 64 * 4);
  float*  CL   = (float*)carve((size_t)NLITS * FMDIM * 4);
  bf16_t* Xl   = (bf16_t*)carve((size_t)NVARS * 384 * 2);
  bf16_t* L2b  = (bf16_t*)carve((size_t)NVARS * FMDIM * 2);
  float*  lpad = (float*)carve((size_t)NVARS * 64 * 4);
  float*  lgts = (float*)carve((size_t)NVARS * 4);
  float*  cv   = (float*)carve((size_t)NCLAUSES * 4);
  float*  csum = (float*)carve(128 * 4);
  float*  csq  = (float*)carve(128 * 4);
  float*  pg   = (float*)carve(NGRAPHS * 4);
  float*  loss = (float*)carve(4);
  bf16_t* frag[9]; float* bpad[9];
  for (int i = 0; i < 9; ++i) {
    int Kt = Ls[i].Kp >> 5, Nt = Ls[i].Np >> 4;
    frag[i] = (bf16_t*)carve((size_t)Nt * Kt * 512 * 2);
    bpad[i] = (float*)carve((size_t)Ls[i].Np * 4);
  }

  const int B = 256;

  // ---- weight packing (every call; deterministic) -------------------------
  for (int i = 0; i < 9; ++i) {
    int Kt = Ls[i].Kp >> 5, Nt = Ls[i].Np >> 4;
    int total = Nt * Kt * 512;
    k_prep_wfrag<<<CDIV(total, B), B, 0, stream>>>(Ls[i].W, frag[i], Ls[i].Kr, Ls[i].Nr, Kt, Nt);
    k_prep_bias<<<CDIV(Ls[i].Np, B), B, 0, stream>>>(Ls[i].B, bpad[i], Ls[i].Nr, Ls[i].Np);
  }

  auto gemm = [&](const bf16_t* X, int li, float* Yf, bf16_t* Yb, int M, int act) {
    int mtiles = M / 16;
    k_gemm_wmma<<<CDIV(mtiles, 4), 128, 0, stream>>>(
        X, frag[li], bpad[li], Yf, Yb, M, Ls[li].Kp, Ls[li].Np, act);
  };

  // ---- init state ---------------------------------------------------------
  k_fill_scalar<<<CDIV(NVARS * FMDIM, B), B, 0, stream>>>(L_a, NVARS * FMDIM, l_init);
  k_fill_scalar<<<CDIV(NCLAUSES * FMDIM, B), B, 0, stream>>>(C_a, NCLAUSES * FMDIM, c_init);
  k_zero<<<CDIV(NVARS, B), B, 0, stream>>>(lgts, NVARS);
  k_zero<<<1, 1, 0, stream>>>(loss, 1);
  // cv0 from logits == 0 (softplus(0)=ln2 per edge)
  k_zero<<<CDIV(NCLAUSES, B), B, 0, stream>>>(cv, NCLAUSES);
  k_edge_cv<<<CDIV(NEDGES, B), B, 0, stream>>>(adj_rows, adj_cols, lgts, cv);

  float* Lc = L_a; float* Ln = L_b;
  float* Cc = C_a; float* Cn = C_b;

  for (int r = 0; r < NROUNDS; ++r) {
    // LC messages (literals -> clauses), 64 features
    k_zero<<<CDIV(NCLAUSES * 64, B), B, 0, stream>>>(LC, NCLAUSES * 64);
    k_edge_lc<<<CDIV(NEDGES * 64, B), B, 0, stream>>>(adj_rows, adj_cols, Lc, LC);

    // Clause update MLP (input: C | exp(-cv) | 0.1*LC, K padded 193->224)
    k_assemble_xc<<<CDIV(NCLAUSES * 224, B), B, 0, stream>>>(Cc, cv, LC, lc_scale, Xc);
    gemm(Xc, 0, nullptr, SA, NCLAUSES, 1);
    gemm(SA, 1, nullptr, SB, NCLAUSES, 1);
    gemm(SB, 2, Cn, nullptr, NCLAUSES, 0);

    // normalize C2
    k_zero<<<1, B, 0, stream>>>(csum, 128);
    k_zero<<<1, B, 0, stream>>>(csq, 128);
    k_colsum<<<CDIV(NCLAUSES, 512), 128, 0, stream>>>(Cn, csum, csq, NCLAUSES, 128);
    k_norm_apply<<<CDIV(NCLAUSES * FMDIM, B), B, 0, stream>>>(Cn, csum, csq, NCLAUSES, 128, nullptr);

    // CL messages (clauses -> literals), 128 features
    k_zero<<<CDIV(NLITS * FMDIM, B), B, 0, stream>>>(CL, NLITS * FMDIM);
    k_edge_cl<<<CDIV(NEDGES * 128, B), B, 0, stream>>>(adj_rows, adj_cols, Cn, CL);

    // Literal update MLP (input: L | 0.1*m1 | 0.1*m2)
    k_assemble_xl<<<CDIV(NVARS * 384, B), B, 0, stream>>>(Lc, CL, cl_scale, Xl);
    gemm(Xl, 3, nullptr, SA, NVARS, 1);
    gemm(SA, 4, nullptr, SB, NVARS, 1);
    gemm(SB, 5, Ln, nullptr, NVARS, 0);

    // normalize L2 (also emit bf16 copy for V_score input)
    k_zero<<<1, B, 0, stream>>>(csum, 128);
    k_zero<<<1, B, 0, stream>>>(csq, 128);
    k_colsum<<<CDIV(NVARS, 512), 128, 0, stream>>>(Ln, csum, csq, NVARS, 128);
    k_norm_apply<<<CDIV(NVARS * FMDIM, B), B, 0, stream>>>(Ln, csum, csq, NVARS, 128, L2b);

    // V_score MLP -> logits (N padded 1->64)
    gemm(L2b, 6, nullptr, SA, NVARS, 1);
    gemm(SA, 7, nullptr, SB, NVARS, 1);
    gemm(SB, 8, lpad, nullptr, NVARS, 0);
    k_extract_logits<<<CDIV(NVARS, B), B, 0, stream>>>(lpad, lgts);

    // cv with new logits (reused as next round's clauses_loss) + loss
    k_zero<<<CDIV(NCLAUSES, B), B, 0, stream>>>(cv, NCLAUSES);
    k_edge_cv<<<CDIV(NEDGES, B), B, 0, stream>>>(adj_rows, adj_cols, lgts, cv);
    k_zero<<<1, B, 0, stream>>>(pg, NGRAPHS);
    k_clause_loss<<<CDIV(NCLAUSES, B), B, 0, stream>>>(cv, gids, pg);
    k_loss_accum<<<1, 1, 0, stream>>>(pg, loss);

    // carry (stop-gradient mix is identity in forward)
    float* t;
    t = Lc; Lc = Ln; Ln = t;
    t = Cc; Cc = Cn; Cn = t;
  }

  k_finalize<<<CDIV(NVARS + 1, B), B, 0, stream>>>(lgts, loss, (float*)d_out);
}
